// HybridAttention_79104707658183
// MI455X (gfx1250) — matrix operand order repaired and checked by reading, not compiled, not run on hardware
//
#include <hip/hip_runtime.h>
#include <hip/hip_bf16.h>

// ---------------- config (matches reference) ----------------
#define DIM   384
#define C1    192
#define WSZ   8
#define NTOK  64           // tokens per window
#define NH    6            // window heads, head_dim 32
#define HD    32
#define CHH   2            // channel heads
#define CHD   96           // channels per channel-head
#define BB    32
#define HH    64
#define WW    64
#define B1    2048         // windows
#define HW    4096
#define M_TOK 131072       // B1*NTOK == B*H*W
#define K3C   576          // 3*C1

typedef __attribute__((ext_vector_type(16))) _Float16 v16h;
typedef __attribute__((ext_vector_type(8)))  _Float16 v8h;
typedef __attribute__((ext_vector_type(8)))  float    v8f;

// ---------------- CDNA5 async global->LDS copy (ASYNCcnt-tracked) ----------------
// GLOBAL_LOAD_ASYNC_TO_LDS_B128: VDST = per-lane LDS byte address, VADDR = 64-bit
// global address (GV mode, SADDR = off). Generic LDS pointers carry the LDS byte
// offset in their low 32 bits (ISA 10.2 aperture mapping), so truncation is valid.
__device__ inline void async_cp16(const _Float16* g, _Float16* l) {
  unsigned lds = (unsigned)(size_t)l;
  asm volatile("global_load_async_to_lds_b128 %0, %1, off"
               :: "v"(lds), "v"(g) : "memory");
}
__device__ inline void async_wait_le3() {
  asm volatile("s_wait_asynccnt 0x3" ::: "memory");
}
__device__ inline void async_wait0() {
  asm volatile("s_wait_asynccnt 0x0" ::: "memory");
}

// ---------------- WMMA fragment helpers (CDNA5 wave32 layouts) ----------------
// A (16x32 f16, row-major, lda elems): lane row = m0+(l&15); half h = l>>4 selects
// K sub-ranges [8h..8h+7] (VGPR0-3) and [16+8h..16+8h+7] (VGPR4-7).
__device__ inline v16h load_a16(const _Float16* A, int lda, int m0, int k0) {
  const int l   = threadIdx.x & 31;
  const int row = m0 + (l & 15);
  const int hh  = l >> 4;
  const _Float16* p = A + (size_t)row * lda + (k0 + 8 * hh);
  v8h lo = *(const v8h*)p;
  v8h hi = *(const v8h*)(p + 16);
  v16h r;
#pragma unroll
  for (int i = 0; i < 8; ++i) { r[i] = lo[i]; r[i + 8] = hi[i]; }
  return r;
}
// B (32x16 f16, row-major K x N): lane col = n0+(l&15); half h = l>>4 selects
// K sub-ranges [8h..8h+7] (elems 0-7) and [16+8h..16+8h+7] (elems 8-15).
__device__ inline v16h load_b16(const _Float16* Bm, int ldb, int k0, int n0) {
  const int l = threadIdx.x & 31;
  const int hh = l >> 4;
  const _Float16* p = Bm + (size_t)(k0 + 8 * hh) * ldb + n0 + (l & 15);
  v16h r;
#pragma unroll
  for (int i = 0; i < 8; ++i) { r[i] = p[(size_t)i * ldb]; r[i + 8] = p[(size_t)(i + 16) * ldb]; }
  return r;
}
__device__ inline v8f wmma_f16(v16h a, v16h b, v8f c) {
  return __builtin_amdgcn_wmma_f32_16x16x32_f16(false, a, false, b, (short)0, c,
                                                false, false);
}

// ---------------- prep kernels ----------------
__global__ void k_cvt(const float* __restrict__ in, _Float16* __restrict__ out, size_t n) {
  size_t i = (size_t)blockIdx.x * blockDim.x + threadIdx.x;
  size_t st = (size_t)gridDim.x * blockDim.x;
  for (; i < n; i += st) out[i] = (_Float16)in[i];
}
// in (R x C) row-major  ->  out (C x R): out[c*R + r] = in[r*C + c]  (f32 -> f16)
__global__ void k_transcvt(const float* __restrict__ in, _Float16* __restrict__ out,
                           int R, int C) {
  int i = blockIdx.x * blockDim.x + threadIdx.x;
  if (i < R * C) {
    int r = i / C, c = i % C;
    out[(size_t)c * R + r] = (_Float16)in[i];
  }
}
// rel-pos bias gather: rpb[h][n][m] = table[rel_index[n][m]][h]
__global__ void k_rpb(const float* __restrict__ tbl, const int* __restrict__ ridx,
                      float* __restrict__ rpb) {
  int i = blockIdx.x * blockDim.x + threadIdx.x;
  if (i < NH * NTOK * NTOK) {
    int h = i / (NTOK * NTOK), nm = i % (NTOK * NTOK);
    rpb[i] = tbl[ridx[nm] * NH + h];
  }
}

// ---------------- async double-buffered GEMM ----------------
// C(MxN) f16 = A(MxK) f16 * B(KxN) f16 + bias.  One block = 128x64 output tile,
// 8 waves of 16x64.  Per k-step: A tile 128x32 (8KB), B tile 32x64 (4KB), staged
// via global_load_async_to_lds_b128, double buffered; wait covers only the
// previous tile (<=3 outstanding per thread) so copy overlaps WMMA.
__global__ void __launch_bounds__(256)
k_gemm(const _Float16* __restrict__ A, const _Float16* __restrict__ Bm,
       const float* __restrict__ bias, _Float16* __restrict__ C,
       int M, int N, int K, int lda, int ldb, int ldc) {
  __shared__ _Float16 As[2][128 * 32];
  __shared__ _Float16 Bs[2][32 * 64];
  const int nblk = N >> 6;
  const int m0 = (blockIdx.x / nblk) << 7;
  const int n0 = (blockIdx.x % nblk) << 6;
  const int t = threadIdx.x;
  const int arow = t >> 1, acol = (t & 1) << 4;     // 2 x 16B per thread
  const int brow = t >> 3, bcol = (t & 7) << 3;     // 1 x 16B per thread
  const _Float16* ga = A + (size_t)(m0 + arow) * lda + acol;
  const _Float16* gb = Bm + (size_t)brow * ldb + n0 + bcol;

  auto stage = [&](int buf, int k0) {
    async_cp16(ga + k0,     &As[buf][arow * 32 + acol]);
    async_cp16(ga + k0 + 8, &As[buf][arow * 32 + acol + 8]);
    async_cp16(gb + (size_t)k0 * ldb, &Bs[buf][brow * 64 + bcol]);
  };

  const int wid = t >> 5;
  const int mw = wid << 4;
  v8f acc[4] = {};
  const int nsteps = K >> 5;
  stage(0, 0);
  for (int s = 0; s < nsteps; ++s) {
    if (s + 1 < nsteps) { stage((s + 1) & 1, (s + 1) << 5); async_wait_le3(); }
    else                { async_wait0(); }
    __syncthreads();
    const _Float16* Ab = As[s & 1];
    const _Float16* Bb = Bs[s & 1];
    v16h a  = load_a16(Ab, 32, mw, 0);
    v16h b0 = load_b16(Bb, 64, 0, 0);
    v16h b1 = load_b16(Bb, 64, 0, 16);
    v16h b2 = load_b16(Bb, 64, 0, 32);
    v16h b3 = load_b16(Bb, 64, 0, 48);
    acc[0] = wmma_f16(a, b0, acc[0]);
    acc[1] = wmma_f16(a, b1, acc[1]);
    acc[2] = wmma_f16(a, b2, acc[2]);
    acc[3] = wmma_f16(a, b3, acc[3]);
    __syncthreads();
  }
  const int l = t & 31;
  const int rowb = m0 + mw + ((l >> 4) << 3);
  const int cb = l & 15;
#pragma unroll
  for (int j = 0; j < 4; ++j) {
    int col = n0 + 16 * j + cb;
    float bv = bias ? bias[col] : 0.0f;
#pragma unroll
    for (int r = 0; r < 8; ++r)
      C[(size_t)(rowb + r) * ldc + col] = (_Float16)(acc[j][r] + bv);
  }
}

// ---------------- fused window attention: one block = (window, head) ----------------
__global__ void k_winattn(const _Float16* __restrict__ qkv, const float* __restrict__ rpb,
                          _Float16* __restrict__ x1o) {
  __shared__ _Float16 qs[NTOK * HD];     // 4 KB
  __shared__ _Float16 kTs[HD * NTOK];    // 4 KB (k transposed -> contiguous B frags)
  __shared__ _Float16 vs[NTOK * HD];     // 4 KB
  __shared__ float    Ss[NTOK * NTOK];   // 16 KB logits
  __shared__ _Float16 Ps[NTOK * NTOK];   // 8 KB probs
  const int b1 = blockIdx.x / NH, head = blockIdx.x % NH;
  const int t = threadIdx.x;             // 0..63, 2 waves
  const _Float16* base = qkv + ((size_t)b1 * NTOK + t) * K3C + head * HD;
#pragma unroll
  for (int j = 0; j < 4; ++j) {          // q, v rows via async copies
    async_cp16(base + 8 * j,          &qs[t * HD + 8 * j]);
    async_cp16(base + 2 * C1 + 8 * j, &vs[t * HD + 8 * j]);
  }
#pragma unroll
  for (int d = 0; d < HD; ++d) kTs[d * NTOK + t] = base[C1 + d];  // transposed k
  async_wait0();
  __syncthreads();

  const int w = t >> 5;
  const int l = t & 31;
  const int rowoff = (l >> 4) << 3;
  const int cb = l & 15;
  const float scale = 0.17677669529663687f;  // 1/sqrt(32)
  // S = scale * q @ k^T + rpb  (64x64, K=32: single WMMA per 16x16 tile)
#pragma unroll
  for (int mi = 0; mi < 2; ++mi) {
    int mt = 2 * w + mi;
    v16h a = load_a16(qs, HD, mt * 16, 0);
#pragma unroll
    for (int nt = 0; nt < 4; ++nt) {
      v8f acc = {};
      acc = wmma_f16(a, load_b16(kTs, NTOK, 0, nt * 16), acc);
#pragma unroll
      for (int r = 0; r < 8; ++r) {
        int row = mt * 16 + rowoff + r, col = nt * 16 + cb;
        Ss[row * NTOK + col] = acc[r] * scale + rpb[head * NTOK * NTOK + row * NTOK + col];
      }
    }
  }
  __syncthreads();
  {  // softmax, one thread per row
    float mx = -1e30f;
    for (int m = 0; m < NTOK; ++m) mx = fmaxf(mx, Ss[t * NTOK + m]);
    float sum = 0.f;
    for (int m = 0; m < NTOK; ++m) {
      float e = __expf(Ss[t * NTOK + m] - mx);
      sum += e;
      Ps[t * NTOK + m] = (_Float16)e;
    }
    float inv = 1.f / sum;
    for (int m = 0; m < NTOK; ++m)
      Ps[t * NTOK + m] = (_Float16)((float)Ps[t * NTOK + m] * inv);
  }
  __syncthreads();
  // O = P @ V  (64x32, K=64)
#pragma unroll
  for (int mi = 0; mi < 2; ++mi) {
    int mt = 2 * w + mi;
    v8f acc[2] = {};
#pragma unroll
    for (int k0 = 0; k0 < NTOK; k0 += 32) {
      v16h a = load_a16(Ps, NTOK, mt * 16, k0);
#pragma unroll
      for (int nt = 0; nt < 2; ++nt)
        acc[nt] = wmma_f16(a, load_b16(vs, HD, k0, nt * 16), acc[nt]);
    }
#pragma unroll
    for (int nt = 0; nt < 2; ++nt)
#pragma unroll
      for (int r = 0; r < 8; ++r) {
        int row = mt * 16 + rowoff + r, col = nt * 16 + cb;
        x1o[((size_t)b1 * NTOK + row) * C1 + head * HD + col] = (_Float16)acc[nt][r];
      }
  }
}

// ---------------- depthwise 3x3 conv (NHWC, SAME, groups = 576) ----------------
__global__ void k_dwconv(const _Float16* __restrict__ in, const float* __restrict__ wdw,
                         _Float16* __restrict__ out) {
  size_t i = (size_t)blockIdx.x * blockDim.x + threadIdx.x;
  if (i >= (size_t)M_TOK * K3C) return;
  int c = (int)(i % K3C);
  size_t p = i / K3C;
  int x = (int)(p & 63), y = (int)((p >> 6) & 63);
  size_t b = p >> 12;
  float acc = 0.f;
#pragma unroll
  for (int dy = 0; dy < 3; ++dy) {
    int yy = y + dy - 1;
    if (yy < 0 || yy >= HH) continue;
#pragma unroll
    for (int dx = 0; dx < 3; ++dx) {
      int xx = x + dx - 1;
      if (xx < 0 || xx >= WW) continue;
      acc += (float)in[(((b << 6) + yy) * (size_t)WW + xx) * K3C + c] * wdw[c * 9 + dy * 3 + dx];
    }
  }
  out[i] = (_Float16)acc;
}

// ---------------- L2 norms over HW per (b,head,channel) for q and k ----------------
__global__ void k_norms(const _Float16* __restrict__ qkv2c, float* __restrict__ rnq,
                        float* __restrict__ rnk) {
  __shared__ float red[64];
  int idx = blockIdx.x;  // b*CHH*CHD + h*CHD + c
  int c = idx % CHD, h = (idx / CHD) % CHH, b = idx / (CHD * CHH);
  int chq = h * CHD + c, chk = C1 + h * CHD + c;
  float sq = 0.f, sk = 0.f;
  for (int p = threadIdx.x; p < HW; p += 64) {
    const _Float16* row = qkv2c + ((size_t)b * HW + p) * K3C;
    float vq = (float)row[chq]; sq += vq * vq;
    float vk = (float)row[chk]; sk += vk * vk;
  }
  red[threadIdx.x] = sq; __syncthreads();
  for (int s = 32; s > 0; s >>= 1) { if (threadIdx.x < s) red[threadIdx.x] += red[threadIdx.x + s]; __syncthreads(); }
  if (threadIdx.x == 0) rnq[idx] = 1.f / fmaxf(sqrtf(red[0]), 1e-12f);
  __syncthreads();
  red[threadIdx.x] = sk; __syncthreads();
  for (int s = 32; s > 0; s >>= 1) { if (threadIdx.x < s) red[threadIdx.x] += red[threadIdx.x + s]; __syncthreads(); }
  if (threadIdx.x == 0) rnk[idx] = 1.f / fmaxf(sqrtf(red[0]), 1e-12f);
}

// ---------------- channel (MDTA) attention: one block = (b, head), 6 waves ----------------
__global__ void k_chattn(const _Float16* __restrict__ qkv2c, const float* __restrict__ rnq,
                         const float* __restrict__ rnk, const float* __restrict__ temp,
                         _Float16* __restrict__ x2o) {
  // union'd LDS: phase1 staging (Qt 12KB + Kp 12KB) overlaps phase2 G (36KB); Pt above.
  __shared__ __align__(32) unsigned char smem[CHD * CHD * 4 + CHD * CHD * 2];  // 54 KB
  float*    G  = (float*)smem;
  _Float16* Pt = (_Float16*)(smem + CHD * CHD * 4);
  _Float16* Qt = (_Float16*)smem;                     // 96 x 64 (channel-major)
  _Float16* Kp = (_Float16*)(smem + CHD * 64 * 2);    // 64 x 96 (pixel-major)
  const int b = blockIdx.x / CHH, h = blockIdx.x % CHH;
  const int t = threadIdx.x;   // 0..191
  const int w = t >> 5;        // wave 0..5
  const size_t gbase = (size_t)b * HW * K3C;
  const int chq = h * CHD, chk = C1 + h * CHD, chv = 2 * C1 + h * CHD;
  const int l = t & 31, rowoff = (l >> 4) << 3, cb = l & 15;

  // Phase 1: G = Q K^T over HW, 64-pixel chunks; wave w owns M-tile w (6 N-tiles)
  v8f acc[6] = {};
  const int sp = t & 63;            // pixel in chunk
  const int sc = (t >> 6) * 32;     // channel chunk base
  for (int p0 = 0; p0 < HW; p0 += 64) {
    __syncthreads();
    const _Float16* rowp = qkv2c + gbase + (size_t)(p0 + sp) * K3C;
#pragma unroll
    for (int j = 0; j < 4; ++j) {
      v8h qv = *(const v8h*)(rowp + chq + sc + 8 * j);
#pragma unroll
      for (int e = 0; e < 8; ++e) Qt[(sc + 8 * j + e) * 64 + sp] = qv[e];
      *(v8h*)&Kp[sp * CHD + sc + 8 * j] = *(const v8h*)(rowp + chk + sc + 8 * j);
    }
    __syncthreads();
#pragma unroll
    for (int ks = 0; ks < 64; ks += 32) {
      v16h a = load_a16(Qt, 64, w * 16, ks);
#pragma unroll
      for (int nt = 0; nt < 6; ++nt)
        acc[nt] = wmma_f16(a, load_b16(Kp, CHD, ks, nt * 16), acc[nt]);
    }
  }
  __syncthreads();
#pragma unroll
  for (int nt = 0; nt < 6; ++nt)
#pragma unroll
    for (int r = 0; r < 8; ++r)
      G[(w * 16 + rowoff + r) * CHD + nt * 16 + cb] = acc[nt][r];
  __syncthreads();
  // Phase 2: scale by temp/(|q||k|), softmax rows, store transposed probs
  if (t < CHD) {
    float rq = rnq[blockIdx.x * CHD + t] * temp[h];
    float mx = -1e30f;
    for (int d = 0; d < CHD; ++d) {
      float g = G[t * CHD + d] * rq * rnk[blockIdx.x * CHD + d];
      G[t * CHD + d] = g;
      mx = fmaxf(mx, g);
    }
    float sum = 0.f;
    for (int d = 0; d < CHD; ++d) { float e = __expf(G[t * CHD + d] - mx); sum += e; G[t * CHD + d] = e; }
    float inv = 1.f / sum;
    for (int d = 0; d < CHD; ++d) Pt[d * CHD + t] = (_Float16)(G[t * CHD + d] * inv);
  }
  __syncthreads();
  // Phase 3: Out(p, c) = sum_d V(p, d) * Pt(d, c); V streams straight from NHWC global
  const _Float16* Av = qkv2c + gbase + chv;
  for (int mt = w; mt < HW / 16; mt += 6) {
    v8f o[6] = {};
#pragma unroll
    for (int ks = 0; ks < CHD; ks += 32) {
      v16h a = load_a16(Av, K3C, mt * 16, ks);
#pragma unroll
      for (int nt = 0; nt < 6; ++nt)
        o[nt] = wmma_f16(a, load_b16(Pt, CHD, ks, nt * 16), o[nt]);
    }
#pragma unroll
    for (int nt = 0; nt < 6; ++nt)
#pragma unroll
      for (int r = 0; r < 8; ++r) {
        int prow = mt * 16 + rowoff + r;
        x2o[((size_t)b * HW + prow) * C1 + h * CHD + nt * 16 + cb] = (_Float16)o[nt][r];
      }
  }
}

// ---------------- final: out = [x1o | window_partition(x2p)] @ w_proj^T + b ----------------
// Same async double-buffered pipeline; window permutation folded into the per-row
// A staging source select; f32 output.
__global__ void __launch_bounds__(256)
k_final(const _Float16* __restrict__ X1o, const _Float16* __restrict__ X2p,
        const _Float16* __restrict__ Bm, const float* __restrict__ bias,
        float* __restrict__ Out) {
  __shared__ _Float16 As[2][128 * 32];
  __shared__ _Float16 Bs[2][32 * 64];
  const int nblk = DIM >> 6;  // 6
  const int m0 = (blockIdx.x / nblk) << 7;
  const int n0 = (blockIdx.x % nblk) << 6;
  const int t = threadIdx.x;
  const int arow = t >> 1, acol = (t & 1) << 4;
  const int brow = t >> 3, bcol = (t & 7) << 3;
  // token (window order) -> pixel order for channel branch, for this staging row
  const int grow = m0 + arow;
  const int b = grow >> 12, r12 = grow & 4095;
  const int b1loc = r12 >> 6, n = r12 & 63;
  const int y = (b1loc >> 3) * 8 + (n >> 3);
  const int x = (b1loc & 7) * 8 + (n & 7);
  const _Float16* pa = X1o + (size_t)grow * C1;
  const _Float16* pb = X2p + (((size_t)b << 12) + (size_t)y * 64 + x) * C1;
  const _Float16* gb = Bm + (size_t)brow * DIM + n0 + bcol;

  auto stage = [&](int buf, int k0) {
    const _Float16* src = (k0 < C1) ? (pa + k0) : (pb + (k0 - C1));
    async_cp16(src + acol,     &As[buf][arow * 32 + acol]);
    async_cp16(src + acol + 8, &As[buf][arow * 32 + acol + 8]);
    async_cp16(gb + (size_t)k0 * DIM, &Bs[buf][brow * 64 + bcol]);
  };

  const int wid = t >> 5;
  const int mw = wid << 4;
  v8f acc[4] = {};
  const int nsteps = DIM >> 5;  // 12
  stage(0, 0);
  for (int s = 0; s < nsteps; ++s) {
    if (s + 1 < nsteps) { stage((s + 1) & 1, (s + 1) << 5); async_wait_le3(); }
    else                { async_wait0(); }
    __syncthreads();
    const _Float16* Ab = As[s & 1];
    const _Float16* Bb = Bs[s & 1];
    v16h a  = load_a16(Ab, 32, mw, 0);
    v16h b0 = load_b16(Bb, 64, 0, 0);
    v16h b1 = load_b16(Bb, 64, 0, 16);
    v16h b2 = load_b16(Bb, 64, 0, 32);
    v16h b3 = load_b16(Bb, 64, 0, 48);
    acc[0] = wmma_f16(a, b0, acc[0]);
    acc[1] = wmma_f16(a, b1, acc[1]);
    acc[2] = wmma_f16(a, b2, acc[2]);
    acc[3] = wmma_f16(a, b3, acc[3]);
    __syncthreads();
  }
  const int l = t & 31;
  const int rowb = m0 + mw + ((l >> 4) << 3);
  const int cb = l & 15;
#pragma unroll
  for (int j = 0; j < 4; ++j) {
    int col = n0 + 16 * j + cb;
    float bv = bias[col];
#pragma unroll
    for (int r = 0; r < 8; ++r)
      Out[(size_t)(rowb + r) * DIM + col] = acc[j][r] + bv;
  }
}

// ---------------- host orchestration ----------------
extern "C" void kernel_launch(void* const* d_in, const int* in_sizes, int n_in,
                              void* d_out, int out_size, void* d_ws, size_t ws_size,
                              hipStream_t stream) {
  (void)in_sizes; (void)n_in; (void)out_size; (void)ws_size;
  const float* x1    = (const float*)d_in[0];
  const float* x2    = (const float*)d_in[1];
  const float* wqkv  = (const float*)d_in[2];   // (576,192)
  const float* bqkv  = (const float*)d_in[3];
  const float* wdw   = (const float*)d_in[4];   // (576,1,3,3)
  const float* wpo   = (const float*)d_in[5];   // (192,192)
  const float* temp  = (const float*)d_in[6];   // (2,1,1)
  const float* btab  = (const float*)d_in[7];   // (225,6)
  const float* wproj = (const float*)d_in[8];   // (384,384)
  const float* bproj = (const float*)d_in[9];
  const int*   ridx  = (const int*)d_in[10];    // (64,64)

  char* ws = (char*)d_ws;
  size_t off = 0;
  auto carve = [&](size_t bytes) { size_t r = off; off += (bytes + 255) & ~(size_t)255; return r; };
  _Float16* x1h    = (_Float16*)(ws + carve((size_t)M_TOK * C1 * 2));
  _Float16* x2h    = (_Float16*)(ws + carve((size_t)M_TOK * C1 * 2));
  _Float16* wqkvT  = (_Float16*)(ws + carve((size_t)C1 * K3C * 2));
  _Float16* wpoT   = (_Float16*)(ws + carve((size_t)C1 * C1 * 2));
  _Float16* wprojT = (_Float16*)(ws + carve((size_t)DIM * DIM * 2));
  float*    rpb    = (float*)   (ws + carve((size_t)NH * NTOK * NTOK * 4));
  float*    rnq    = (float*)   (ws + carve((size_t)BB * CHH * CHD * 4));
  float*    rnk    = (float*)   (ws + carve((size_t)BB * CHH * CHD * 4));
  _Float16* qkv1   = (_Float16*)(ws + carve((size_t)M_TOK * K3C * 2));
  _Float16* qkv2   = (_Float16*)(ws + carve((size_t)M_TOK * K3C * 2));
  _Float16* qkv2c  = (_Float16*)(ws + carve((size_t)M_TOK * K3C * 2));
  _Float16* x1o    = (_Float16*)(ws + carve((size_t)M_TOK * C1 * 2));
  _Float16* x2o    = (_Float16*)(ws + carve((size_t)M_TOK * C1 * 2));
  _Float16* x2p    = (_Float16*)(ws + carve((size_t)M_TOK * C1 * 2));

  // prep: f32->f16 activations, K-major f16 weights, rel-pos bias table
  k_cvt<<<4096, 256, 0, stream>>>(x1, x1h, (size_t)M_TOK * C1);
  k_cvt<<<4096, 256, 0, stream>>>(x2, x2h, (size_t)M_TOK * C1);
  k_transcvt<<<(K3C * C1 + 255) / 256, 256, 0, stream>>>(wqkv, wqkvT, K3C, C1);
  k_transcvt<<<(C1 * C1 + 255) / 256, 256, 0, stream>>>(wpo, wpoT, C1, C1);
  k_transcvt<<<(DIM * DIM + 255) / 256, 256, 0, stream>>>(wproj, wprojT, DIM, DIM);
  k_rpb<<<(NH * NTOK * NTOK + 255) / 256, 256, 0, stream>>>(btab, ridx, rpb);

  // branch 1: qkv GEMM + fused window attention
  k_gemm<<<(M_TOK / 128) * (K3C / 64), 256, 0, stream>>>(
      x1h, wqkvT, bqkv, qkv1, M_TOK, K3C, C1, C1, K3C, K3C);
  k_winattn<<<B1 * NH, 64, 0, stream>>>(qkv1, rpb, x1o);

  // branch 2: qkv GEMM, depthwise conv, norms, channel attention, proj_out
  k_gemm<<<(M_TOK / 128) * (K3C / 64), 256, 0, stream>>>(
      x2h, wqkvT, bqkv, qkv2, M_TOK, K3C, C1, C1, K3C, K3C);
  k_dwconv<<<(int)(((size_t)M_TOK * K3C + 255) / 256), 256, 0, stream>>>(qkv2, wdw, qkv2c);
  k_norms<<<BB * CHH * CHD, 64, 0, stream>>>(qkv2c, rnq, rnk);
  k_chattn<<<BB * CHH, 192, 0, stream>>>(qkv2c, rnq, rnk, temp, x2o);
  k_gemm<<<(M_TOK / 128) * (C1 / 64), 256, 0, stream>>>(
      x2o, wpoT, nullptr, x2p, M_TOK, C1, C1, C1, C1, C1);

  // fuse: concat (with window-partition gather) + output projection, f32 out
  k_final<<<(M_TOK / 128) * (DIM / 64), 256, 0, stream>>>(
      x1o, x2p, wprojT, bproj, (float*)d_out);
}
